// FlaxLLaMAAttention_6889127542833
// MI455X (gfx1250) — matrix operand order repaired and checked by reading, not compiled, not run on hardware
//
#include <hip/hip_runtime.h>

// ---------------------------------------------------------------------------
// LLaMA attention block (B=1,S=2048,D=2048,H=32,KVH=8,HD=64) for gfx1250.
// bf16 WMMA (v_wmma_f32_16x16x32_bf16) + double-buffered TDM
// (tensor_load_to_lds, s_wait_tensorcnt 1) staging + global_prefetch.
// ---------------------------------------------------------------------------

typedef __attribute__((ext_vector_type(16))) __bf16 v16bf;
typedef __attribute__((ext_vector_type(8)))  __bf16 v8bf;
typedef __attribute__((ext_vector_type(8)))  float  v8f;
typedef __attribute__((ext_vector_type(4)))  unsigned u32x4;
typedef __attribute__((ext_vector_type(8)))  unsigned u32x8;

#define SHUF16(lo, hi) __builtin_shufflevector((lo), (hi), 0,1,2,3,4,5,6,7,8,9,10,11,12,13,14,15)

constexpr int S   = 2048;
constexpr int D   = 2048;
constexpr int H   = 32;
constexpr int KVH = 8;
constexpr int HD  = 64;
constexpr int NQ  = H * HD;    // 2048
constexpr int NKV = KVH * HD;  // 512

__device__ __forceinline__ unsigned short f2bf(float f) {
  unsigned int u = __float_as_uint(f);
  u += 0x7fffu + ((u >> 16) & 1u);   // round-to-nearest-even
  return (unsigned short)(u >> 16);
}

// ------------------------- elementwise helpers ------------------------------

__global__ void k_conv(const float* __restrict__ src,
                       unsigned short* __restrict__ dst, int n) {
  int i = blockIdx.x * blockDim.x + threadIdx.x;
  if (i < n) dst[i] = f2bf(src[i]);
}

// src is [K x N] f32 row-major; dst is [N x K] bf16 (i.e. src^T)
__global__ void k_transpose(const float* __restrict__ src,
                            unsigned short* __restrict__ dst, int K, int N) {
  int i = blockIdx.x * blockDim.x + threadIdx.x;
  if (i >= K * N) return;
  int n = i % N, k = i / N;
  dst[(size_t)n * K + k] = f2bf(src[(size_t)k * N + n]);
}

// RoPE: x is [S, nh*HD] f32; out is [nh][S][HD] bf16 (post-rotation)
__global__ void k_rope(const float* __restrict__ x,
                       unsigned short* __restrict__ o, int nh) {
  int i = blockIdx.x * blockDim.x + threadIdx.x;
  int total = S * nh * (HD / 2);
  if (i >= total) return;
  int p  = i & (HD / 2 - 1);
  int sh = i >> 5;            // HD/2 == 32
  int h  = sh % nh;
  int s  = sh / nh;
  size_t src = (size_t)s * nh * HD + (size_t)h * HD + 2 * p;
  float a = x[src], b = x[src + 1];
  float freq = __powf(10000.0f, -(2.0f * (float)p) / (float)HD);
  float sn, cs;
  __sincosf((float)s * freq, &sn, &cs);
  size_t dst = ((size_t)h * S + s) * HD + 2 * p;
  o[dst]     = f2bf(a * cs - b * sn);
  o[dst + 1] = f2bf(a * sn + b * cs);
}

// V: [S, NKV] f32  ->  Vt: [KVH][HD][S] bf16
__global__ void k_vtrans(const float* __restrict__ v,
                         unsigned short* __restrict__ vt) {
  int i = blockIdx.x * blockDim.x + threadIdx.x;
  if (i >= KVH * HD * S) return;
  int s    = i & (S - 1);
  int rest = i >> 11;         // S == 2048
  int d    = rest & (HD - 1);
  int g    = rest >> 6;
  vt[((size_t)g * HD + d) * S + s] = f2bf(v[(size_t)s * NKV + g * HD + d]);
}

// ------------------------- bf16 WMMA fragment loads -------------------------
// A 16x32 bf16 layout: lane m (0..15) holds M=m, K in {hb*8+0..7, hb*8+16..23}
// where hb = lane>>4.  B 32x16 layout: lane n holds column n, same K set as A.
// Both reduce to per-lane row loads from row-major storage.

__device__ __forceinline__ v16bf load_a_frag(const __bf16* rowPtr) {
  v8bf lo = *reinterpret_cast<const v8bf*>(rowPtr);
  v8bf hi = *reinterpret_cast<const v8bf*>(rowPtr + 16);
  return SHUF16(lo, hi);
}
__device__ __forceinline__ v16bf load_b_frag(const __bf16* rowPtr) {
  v8bf lo = *reinterpret_cast<const v8bf*>(rowPtr);
  v8bf hi = *reinterpret_cast<const v8bf*>(rowPtr + 16);
  return SHUF16(lo, hi);
}

// ------------------------- TDM slab loader ----------------------------------
// DMA a 64-row x 32-col bf16 slab of Bt (row stride K elements) into LDS.
// D# group0: count=1 | lds_addr | 57-bit global_addr | type=2 ("image").
// D# group1: data_size=2B; tensor = K x N rows; tile = 32 x 64; stride0 = K.

__device__ __forceinline__ void tdm_load_slab(const unsigned short* Bt,
                                              unsigned ldsOff, int nBase,
                                              int k, int K, int N) {
  unsigned long long ga =
      (unsigned long long)(size_t)Bt + ((size_t)nBase * K + k) * 2ull;
  u32x4 g0 = {1u, ldsOff, (unsigned)ga, (unsigned)(ga >> 32) | 0x80000000u};
  u32x8 g1 = {0x00010000u,
              ((unsigned)K & 0xffffu) << 16,
              ((unsigned)K >> 16) | (((unsigned)N & 0xffffu) << 16),
              ((unsigned)N >> 16) | (32u << 16),
              64u,
              (unsigned)K,
              0u, 0u};
  asm volatile("tensor_load_to_lds %0, %1" :: "s"(g0), "s"(g1) : "memory");
}

// ------------------------- GEMM: C[M,N] = A[M,K] * Bt[N,K]^T ----------------
// 128 threads = 4 waves; wave tile 32(M) x 64(N); block tile 128 x 64.
// Double-buffered TDM: slab i+1 DMA overlaps compute of slab i; wave 0 waits
// s_wait_tensorcnt 1 (TDM ops complete in order) before publishing a slab.

__global__ __launch_bounds__(128) void k_gemm(
    const unsigned short* __restrict__ A, const unsigned short* __restrict__ Bt,
    float* __restrict__ C, int M, int N, int K) {
  __shared__ __align__(16) unsigned short Bs[2][64 * 32];

  const int w    = threadIdx.x >> 5;
  const int lane = threadIdx.x & 31;
  const int m    = lane & 15;
  const int hb   = lane >> 4;
  const int mBase = blockIdx.y * 128 + w * 32;
  const int nBase = blockIdx.x * 64;
  (void)M;

  const __bf16* Ab = reinterpret_cast<const __bf16*>(A);
  // Flat->LDS address truncation (ISA 10.2): keep the low 32 bits.
  const unsigned ldsOff0 = (unsigned)(size_t)(&Bs[0][0]);
  const unsigned ldsOff1 = (unsigned)(size_t)(&Bs[1][0]);

  v8f zero = {0.f, 0.f, 0.f, 0.f, 0.f, 0.f, 0.f, 0.f};
  v8f acc[8];
#pragma unroll
  for (int t = 0; t < 8; ++t) acc[t] = zero;

  const int nsteps = K >> 5;
  if (w == 0) tdm_load_slab(Bt, ldsOff0, nBase, 0, K, N);

  for (int i = 0; i < nsteps; ++i) {
    const int cur = i & 1;
    const int k   = i << 5;
    __syncthreads();  // all waves done reading slab[cur^1] (prev iteration)
    if (w == 0) {
      if (i + 1 < nsteps) {
        tdm_load_slab(Bt, cur ? ldsOff0 : ldsOff1, nBase, k + 32, K, N);
        __builtin_amdgcn_s_wait_tensorcnt(1);  // older DMA (slab[cur]) done
      } else {
        __builtin_amdgcn_s_wait_tensorcnt(0);
      }
    }
    __syncthreads();  // slab[cur] published to all 4 waves

    const __bf16* ap0 = Ab + (size_t)(mBase + m) * K + k + hb * 8;
    const __bf16* ap1 = Ab + (size_t)(mBase + 16 + m) * K + k + hb * 8;
    v16bf a0 = load_a_frag(ap0);
    v16bf a1 = load_a_frag(ap1);
    __builtin_prefetch(ap0 + 64, 0, 1);  // global_prefetch: two k-steps ahead
    __builtin_prefetch(ap1 + 64, 0, 1);

    const __bf16* Bsb = reinterpret_cast<const __bf16*>(&Bs[cur][0]);
#pragma unroll
    for (int t = 0; t < 4; ++t) {
      const __bf16* bp = Bsb + (t * 16 + m) * 32 + hb * 8;
      v16bf b = load_b_frag(bp);
      acc[t]     = __builtin_amdgcn_wmma_f32_16x16x32_bf16(
          false, a0, false, b, (short)0, acc[t], false, false);
      acc[4 + t] = __builtin_amdgcn_wmma_f32_16x16x32_bf16(
          false, a1, false, b, (short)0, acc[4 + t], false, false);
    }
  }

#pragma unroll
  for (int rg = 0; rg < 2; ++rg) {
#pragma unroll
    for (int t = 0; t < 4; ++t) {
#pragma unroll
      for (int r = 0; r < 8; ++r) {
        C[(size_t)(mBase + rg * 16 + hb * 8 + r) * N + nBase + t * 16 + m] =
            acc[rg * 4 + t][r];
      }
    }
  }
}

// ------------------------- Flash attention ----------------------------------
// One wave per 16-row query tile per head.  Online softmax over kv steps of
// 32 keys.  Scores: WMMA with K=HD=64.  P (C-layout) -> A-layout transpose via
// per-wave LDS staging + s_wait_dscnt.  Output written directly as bf16.

__global__ __launch_bounds__(128) void k_flash(
    const unsigned short* __restrict__ qbm, const unsigned short* __restrict__ kbm,
    const unsigned short* __restrict__ vtm, unsigned short* __restrict__ attnb) {
  __shared__ __align__(16) unsigned short pbuf[4 * 16 * 32];

  const int w    = threadIdx.x >> 5;
  const int lane = threadIdx.x & 31;
  const int m    = lane & 15;
  const int hb   = lane >> 4;
  const int qbase = (blockIdx.x * 4 + w) * 16;
  const int h     = blockIdx.y;
  const int g     = h >> 2;  // GQA: n_rep = H/KVH = 4

  const __bf16* Q  = reinterpret_cast<const __bf16*>(qbm) + ((size_t)h * S + qbase) * HD;
  const __bf16* Kb = reinterpret_cast<const __bf16*>(kbm) + (size_t)g * S * HD;
  const __bf16* Vt = reinterpret_cast<const __bf16*>(vtm) + (size_t)g * HD * S;

  const __bf16* qp = Q + (size_t)m * HD + hb * 8;
  v16bf a0 = load_a_frag(qp);
  v16bf a1 = load_a_frag(qp + 32);

  v8f zero = {0.f, 0.f, 0.f, 0.f, 0.f, 0.f, 0.f, 0.f};
  v8f acc[4];
#pragma unroll
  for (int t = 0; t < 4; ++t) acc[t] = zero;
  float mr[8], lr[8];
#pragma unroll
  for (int r = 0; r < 8; ++r) { mr[r] = -3.0e38f; lr[r] = 0.0f; }

  unsigned short* pb = pbuf + w * (16 * 32);
  const int steps = (qbase + 47) >> 5;  // covers kv 0 .. qbase+15 (causal)
  const float qk_scale = 0.125f;        // 1/sqrt(64)

  for (int s0 = 0; s0 < steps; ++s0) {
    const int kv0 = s0 << 5;

    v8f st[2];
#pragma unroll
    for (int t = 0; t < 2; ++t) {
      const __bf16* kp = Kb + (size_t)(kv0 + t * 16 + m) * HD;
      v16bf b0 = load_b_frag(kp + hb * 8);
      v16bf b1 = load_b_frag(kp + 32 + hb * 8);
      st[t] = __builtin_amdgcn_wmma_f32_16x16x32_bf16(
          false, a0, false, b0, (short)0, zero, false, false);
      st[t] = __builtin_amdgcn_wmma_f32_16x16x32_bf16(
          false, a1, false, b1, (short)0, st[t], false, false);
    }

    const bool needMask = (kv0 + 31 > qbase);  // uniform: only diagonal step

#pragma unroll
    for (int r = 0; r < 8; ++r) {
      const int row = qbase + hb * 8 + r;
      float s0v = st[0][r] * qk_scale;
      float s1v = st[1][r] * qk_scale;
      if (needMask) {
        if (kv0 + m > row)      s0v = -3.0e38f;
        if (kv0 + 16 + m > row) s1v = -3.0e38f;
      }
      float mx = fmaxf(s0v, s1v);
#pragma unroll
      for (int o = 1; o < 16; o <<= 1) mx = fmaxf(mx, __shfl_xor(mx, o, 32));
      const float nm   = fmaxf(mr[r], mx);
      const float corr = __expf(mr[r] - nm);
      const float p0   = __expf(s0v - nm);
      const float p1   = __expf(s1v - nm);
      float ps = p0 + p1;
#pragma unroll
      for (int o = 1; o < 16; o <<= 1) ps += __shfl_xor(ps, o, 32);
      lr[r] = lr[r] * corr + ps;
      mr[r] = nm;
#pragma unroll
      for (int t = 0; t < 4; ++t) acc[t][r] *= corr;
      pb[(hb * 8 + r) * 32 + m]      = f2bf(p0);
      pb[(hb * 8 + r) * 32 + 16 + m] = f2bf(p1);
    }

    asm volatile("s_wait_dscnt 0" ::: "memory");

    const __bf16* pp = reinterpret_cast<const __bf16*>(pb) + m * 32 + hb * 8;
    v16bf pa = load_a_frag(pp);

#pragma unroll
    for (int t = 0; t < 4; ++t) {
      const __bf16* vp = Vt + (size_t)(t * 16 + m) * S + kv0 + hb * 8;
      v16bf vb = load_b_frag(vp);
      acc[t] = __builtin_amdgcn_wmma_f32_16x16x32_bf16(
          false, pa, false, vb, (short)0, acc[t], false, false);
    }
  }

  // normalize, store bf16 [S, NQ] directly (input to output projection)
#pragma unroll
  for (int r = 0; r < 8; ++r) {
    const float inv = 1.0f / lr[r];
#pragma unroll
    for (int t = 0; t < 4; ++t) {
      attnb[(size_t)(qbase + hb * 8 + r) * NQ + h * HD + t * 16 + m] =
          f2bf(acc[t][r] * inv);
    }
  }
}

// ------------------------- host-side orchestration --------------------------

extern "C" void kernel_launch(void* const* d_in, const int* in_sizes, int n_in,
                              void* d_out, int out_size, void* d_ws,
                              size_t ws_size, hipStream_t stream) {
  const float* hs = (const float*)d_in[0];
  const float* wq = (const float*)d_in[1];
  const float* wk = (const float*)d_in[2];
  const float* wv = (const float*)d_in[3];
  const float* wo = (const float*)d_in[4];
  float* out = (float*)d_out;
  (void)in_sizes; (void)n_in; (void)out_size; (void)ws_size;

  char* ws = (char*)d_ws;
  size_t off = 0;
  auto take = [&](size_t bytes) -> void* {
    void* p = ws + off;
    off += (bytes + 255) & ~(size_t)255;
    return p;
  };

  unsigned short* hsb   = (unsigned short*)take((size_t)S * D * 2);
  unsigned short* wqT   = (unsigned short*)take((size_t)NQ * D * 2);
  unsigned short* wkT   = (unsigned short*)take((size_t)NKV * D * 2);
  unsigned short* wvT   = (unsigned short*)take((size_t)NKV * D * 2);
  unsigned short* woT   = (unsigned short*)take((size_t)D * NQ * 2);
  float*          qf    = (float*)take((size_t)S * NQ * 4);
  float*          kf    = (float*)take((size_t)S * NKV * 4);
  float*          vf    = (float*)take((size_t)S * NKV * 4);
  unsigned short* qb    = (unsigned short*)take((size_t)H * S * HD * 2);
  unsigned short* kbp   = (unsigned short*)take((size_t)KVH * S * HD * 2);
  unsigned short* vtp   = (unsigned short*)take((size_t)KVH * HD * S * 2);
  unsigned short* attnb = (unsigned short*)take((size_t)S * NQ * 2);

  const int TB = 256;
  // bf16 conversions / transposes
  k_conv<<<(S * D + TB - 1) / TB, TB, 0, stream>>>(hs, hsb, S * D);
  k_transpose<<<(D * NQ + TB - 1) / TB, TB, 0, stream>>>(wq, wqT, D, NQ);
  k_transpose<<<(D * NKV + TB - 1) / TB, TB, 0, stream>>>(wk, wkT, D, NKV);
  k_transpose<<<(D * NKV + TB - 1) / TB, TB, 0, stream>>>(wv, wvT, D, NKV);
  k_transpose<<<(NQ * D + TB - 1) / TB, TB, 0, stream>>>(wo, woT, NQ, D);
  // projections (M=2048 rows, block tile 128 rows)
  k_gemm<<<dim3(NQ / 64, S / 128), 128, 0, stream>>>(hsb, wqT, qf, S, NQ, D);
  k_gemm<<<dim3(NKV / 64, S / 128), 128, 0, stream>>>(hsb, wkT, kf, S, NKV, D);
  k_gemm<<<dim3(NKV / 64, S / 128), 128, 0, stream>>>(hsb, wvT, vf, S, NKV, D);
  // RoPE + V transpose into attention-friendly bf16 layouts
  k_rope<<<(S * H * (HD / 2) + TB - 1) / TB, TB, 0, stream>>>(qf, qb, H);
  k_rope<<<(S * KVH * (HD / 2) + TB - 1) / TB, TB, 0, stream>>>(kf, kbp, KVH);
  k_vtrans<<<(KVH * HD * S + TB - 1) / TB, TB, 0, stream>>>(vf, vtp);
  // causal flash attention (writes bf16)
  k_flash<<<dim3(S / 64, H), 128, 0, stream>>>(qb, kbp, vtp, attnb);
  // output projection
  k_gemm<<<dim3(D / 64, S / 128), 128, 0, stream>>>(attnb, woT, out, S, D, NQ);
}
